// GPTLanguageModel_14714557956572
// MI455X (gfx1250) — matrix operand order repaired and checked by reading, not compiled, not run on hardware
//
#include <hip/hip_runtime.h>
#include <hip/hip_bf16.h>
#include <stdint.h>

#define DI __device__ __forceinline__

typedef __attribute__((ext_vector_type(16))) __bf16          v16bf;
typedef __attribute__((ext_vector_type(16))) unsigned short  v16u;
typedef __attribute__((ext_vector_type(8)))  float           v8f;
typedef __attribute__((ext_vector_type(4)))  unsigned int    u32x4;
typedef __attribute__((ext_vector_type(8)))  int             i32x8;
typedef __attribute__((ext_vector_type(4)))  int             i32x4;

#if __has_builtin(__builtin_amdgcn_tensor_load_to_lds)
#define HAVE_TDM 1
#else
#define HAVE_TDM 0
#endif

struct U128x2 { uint4 lo, hi; };

DI v16bf frag_from(uint4 lo, uint4 hi) {
  U128x2 t{lo, hi};
  return __builtin_bit_cast(v16bf, t);
}

DI unsigned short f2bf(float x) {           // fp32 -> bf16, round-nearest-even
  unsigned u = __float_as_uint(x);
  unsigned r = u + 0x7FFFu + ((u >> 16) & 1u);
  return (unsigned short)(r >> 16);
}

DI v8f wmma_bf16(v16bf a, v16bf b, v8f c) {
  // D = A(16x32 bf16) * B(32x16 bf16) + C(16x16 f32)
  return __builtin_amdgcn_wmma_f32_16x16x32_bf16(false, a, false, b, (short)0, c,
                                                 false, false);
}

// ---- model constants -------------------------------------------------------
constexpr int Bc = 2, Tc = 2048, Ec = 512, Hc = 8, HDc = 64, Lc = 6, Vc = 32000;
constexpr int Mc = Bc * Tc;                 // 4096 token rows

// ---- embedding: x = tok_emb[idx] + pos_emb ---------------------------------
__global__ void embed_kernel(const int* __restrict__ idx,
                             const float* __restrict__ tok,
                             const float* __restrict__ pos,
                             float* __restrict__ x) {
  int g   = blockIdx.x * blockDim.x + threadIdx.x;   // over M*E/4 float4's
  int row = g >> 7;                                  // E/4 = 128 chunks per row
  int c   = g & 127;
  int t   = row & (Tc - 1);
  int tokid = idx[row];
  float4 a = ((const float4*)tok)[(size_t)tokid * 128 + c];
  float4 p = ((const float4*)pos)[(size_t)t * 128 + c];
  float4 o;
  o.x = a.x + p.x; o.y = a.y + p.y; o.z = a.z + p.z; o.w = a.w + p.w;
  ((float4*)x)[(size_t)row * 128 + c] = o;
}

// ---- layernorm: fp32 row -> bf16 row (1 wave / row, 8 waves / block) -------
__global__ __launch_bounds__(256) void ln_kernel(const float* __restrict__ x,
                                                 const float* __restrict__ sc,
                                                 const float* __restrict__ bi,
                                                 unsigned short* __restrict__ h) {
  int lane = threadIdx.x & 31, w = threadIdx.x >> 5;
  int row  = blockIdx.x * 8 + w;
  const float4* xr = (const float4*)(x + (size_t)row * Ec);
  float4 v[4];
  float sum = 0.0f;
#pragma unroll
  for (int i = 0; i < 4; ++i) {
    v[i] = xr[i * 32 + lane];
    sum += v[i].x + v[i].y + v[i].z + v[i].w;
  }
#pragma unroll
  for (int d = 16; d >= 1; d >>= 1) sum += __shfl_xor(sum, d, 32);
  float mu = sum * (1.0f / Ec);
  float var = 0.0f;
#pragma unroll
  for (int i = 0; i < 4; ++i) {
    float a = v[i].x - mu, b = v[i].y - mu, c = v[i].z - mu, d = v[i].w - mu;
    var += a * a + b * b + c * c + d * d;
  }
#pragma unroll
  for (int d = 16; d >= 1; d >>= 1) var += __shfl_xor(var, d, 32);
  float rstd = rsqrtf(var * (1.0f / Ec) + 1e-5f);
  ushort4* hr = (ushort4*)(h + (size_t)row * Ec);
#pragma unroll
  for (int i = 0; i < 4; ++i) {
    float4 s = ((const float4*)sc)[i * 32 + lane];
    float4 b = ((const float4*)bi)[i * 32 + lane];
    ushort4 o;
    o.x = f2bf((v[i].x - mu) * rstd * s.x + b.x);
    o.y = f2bf((v[i].y - mu) * rstd * s.y + b.y);
    o.z = f2bf((v[i].z - mu) * rstd * s.z + b.z);
    o.w = f2bf((v[i].w - mu) * rstd * s.w + b.w);
    hr[i * 32 + lane] = o;
  }
}

// ---- weight convert + transpose: fp32 W[K,N] -> bf16 Wt[N,K] ---------------
__global__ __launch_bounds__(256) void convT_kernel(const float* __restrict__ W,
                                                    unsigned short* __restrict__ Wt,
                                                    int K, int N) {
  __shared__ float tile[32][33];
  int tx = threadIdx.x & 31, ty = threadIdx.x >> 5;    // 32 x 8
  int n0 = blockIdx.x * 32, k0 = blockIdx.y * 32;
#pragma unroll
  for (int r = 0; r < 4; ++r)
    tile[ty + r * 8][tx] = W[(size_t)(k0 + ty + r * 8) * N + n0 + tx];
  __syncthreads();
#pragma unroll
  for (int r = 0; r < 4; ++r) {
    int nr = ty + r * 8;
    Wt[(size_t)(n0 + nr) * K + k0 + tx] = f2bf(tile[tx][nr]);
  }
}

// ---- bf16 WMMA GEMM: C[M,N] = A[M,K] * Wt[N,K]^T (+bias)(+relu)(+res) ------
// block tile 128x128, 8 waves, each wave 64x32 = 4x2 subtiles of 16x16.
// Weight tile (Bs) is staged by the Tensor Data Mover when available; its
// pad feature (64B data + 16B pad per row) reproduces the 40-ushort LDS rows.
template <bool BIAS, bool RELU, bool RES, bool OUTF, bool OUTB>
__global__ __launch_bounds__(256) void gemm_kernel(
    const unsigned short* __restrict__ A, const unsigned short* __restrict__ Bt,
    const float* __restrict__ bias, const float* __restrict__ res,
    float* __restrict__ Cf, unsigned short* __restrict__ Cb,
    int M, int N, int K) {
  __shared__ __align__(16) unsigned short As[128][40];   // +8 pad, rows 16B-aligned
  __shared__ __align__(16) unsigned short Bs[128][40];   // stored [n][k]
  const int tid  = threadIdx.x;
  const int lane = tid & 31, w = tid >> 5;
  const int wm = w & 1, wn = w >> 1;                     // 2 x 4 wave grid
  const int m0 = blockIdx.y * 128, n0 = blockIdx.x * 128;
  const int l15 = lane & 15;
  const int m8  = (lane & 16) ? 8 : 0;

  v8f z = {0.f, 0.f, 0.f, 0.f, 0.f, 0.f, 0.f, 0.f};
  v8f acc[4][2];
#pragma unroll
  for (int i = 0; i < 4; ++i)
#pragma unroll
    for (int j = 0; j < 2; ++j) acc[i][j] = z;

  for (int kk = 0; kk < K; kk += 32) {
#if HAVE_TDM
    if (w == 0) {
      // ---- TDM descriptor: 2D tile 32(k) x 128(n) of 2-byte elements ------
      unsigned long long ga =
          (unsigned long long)(uintptr_t)(Bt + (size_t)n0 * K + kk);
      unsigned ldsB = (unsigned)(uintptr_t)(&Bs[0][0]);
      u32x4 g0;
      g0[0] = 1u;                                        // count=1 (user D#)
      g0[1] = ldsB;                                      // lds_addr
      g0[2] = (unsigned)(ga & 0xFFFFFFFFu);              // global_addr[31:0]
      g0[3] = (unsigned)((ga >> 32) & 0x01FFFFFFu)       // global_addr[56:32]
              | (2u << 30);                              // type = 2 ("image")
      i32x8 g1;
      g1[0] = (int)((1u << 16)      // data_size = 2 bytes
                    | (1u << 20)    // pad_enable
                    | (3u << 22)    // pad_interval: every 16 DW (64B row)
                    | (3u << 25));  // pad_amount: 4 DW (16B) -> 80B LDS rows
      g1[1] = (int)(((unsigned)K & 0xFFFFu) << 16);                    // td0 lo
      g1[2] = (int)((((unsigned)K >> 16) & 0xFFFFu) |
                    (((unsigned)N & 0xFFFFu) << 16));                  // td0 hi, td1 lo
      g1[3] = (int)((((unsigned)N >> 16) & 0xFFFFu) | (32u << 16));    // td1 hi, tile_dim0
      g1[4] = (int)128u;                                 // tile_dim1 (tile_dim2=0)
      g1[5] = (int)(unsigned)K;                          // tensor_dim0_stride lo
      g1[6] = 0;                                         // stride hi, dim1_stride lo
      g1[7] = 0;
      i32x4 gz = {0, 0, 0, 0};
#if defined(__clang_major__) && (__clang_major__ >= 23)
      i32x8 gz8 = {0, 0, 0, 0, 0, 0, 0, 0};
      __builtin_amdgcn_tensor_load_to_lds(g0, g1, gz, gz, gz8, 0);
#else
      __builtin_amdgcn_tensor_load_to_lds(g0, g1, gz, gz, 0);
#endif
    }
#endif
#pragma unroll
    for (int p = 0; p < 2; ++p) {                        // A: 512 uint4 / 256 thr
      int gi = p * 256 + tid;
      int r = gi >> 2, c = (gi & 3) * 8;
      *(uint4*)&As[r][c] = *(const uint4*)(A + (size_t)(m0 + r) * K + kk + c);
      if (kk + 32 < K)                                   // -> global_prefetch_b8
        __builtin_prefetch(A + (size_t)(m0 + r) * K + kk + 32 + c, 0, 1);
#if !HAVE_TDM
      *(uint4*)&Bs[r][c] = *(const uint4*)(Bt + (size_t)(n0 + r) * K + kk + c);
#endif
    }
#if HAVE_TDM
    if (w == 0) asm volatile("s_wait_tensorcnt 0x0" ::: "memory");
#endif
    __syncthreads();

    v16bf afr[4], bfr[2];
    const int ab = (lane & 16) ? 16 : 0;                 // A-frag K base (contiguous)
#pragma unroll
    for (int i = 0; i < 4; ++i) {
      const unsigned short* pr = &As[wm * 64 + i * 16 + l15][ab];
      afr[i] = frag_from(*(const uint4*)pr, *(const uint4*)(pr + 8));
    }
    const int kb = (lane & 16) ? 16 : 0;                 // B-frag K base
#pragma unroll
    for (int j = 0; j < 2; ++j) {
      const unsigned short* pr = &Bs[wn * 32 + j * 16 + l15][kb];
      bfr[j] = frag_from(*(const uint4*)pr, *(const uint4*)(pr + 8));
    }
#pragma unroll
    for (int i = 0; i < 4; ++i)
#pragma unroll
      for (int j = 0; j < 2; ++j)
        acc[i][j] = wmma_bf16(afr[i], bfr[j], acc[i][j]);
    __syncthreads();
  }

#pragma unroll
  for (int i = 0; i < 4; ++i) {
#pragma unroll
    for (int j = 0; j < 2; ++j) {
      int n = n0 + wn * 32 + j * 16 + l15;
      float bv = BIAS ? bias[n] : 0.0f;
#pragma unroll
      for (int v = 0; v < 8; ++v) {
        int m = m0 + wm * 64 + i * 16 + v + m8;
        float val = acc[i][j][v];
        if (BIAS) val += bv;
        if (RELU) val = fmaxf(val, 0.0f);
        if (RES)  val += res[(size_t)m * N + n];
        if (OUTF) Cf[(size_t)m * N + n] = val;
        if (OUTB) Cb[(size_t)m * N + n] = f2bf(val);
      }
    }
  }
}

// ---- flash attention: one wave per (b, h, 16-row q tile) -------------------
__global__ __launch_bounds__(256) void attn_kernel(
    const unsigned short* __restrict__ q, const unsigned short* __restrict__ k,
    const unsigned short* __restrict__ vv, unsigned short* __restrict__ att) {
  __shared__ __align__(16) unsigned short Pbuf[8][16 * 32];  // per-wave P tile
  const int lane = threadIdx.x & 31, w = threadIdx.x >> 5;
  const int task = blockIdx.x * 8 + w;      // 2048 tasks total
  const int qt = task & 127;                // q tile (T/16 = 128)
  const int h  = (task >> 7) & 7;
  const int b  = task >> 10;
  const int l15 = lane & 15;
  const int m8  = (lane & 16) ? 8 : 0;
  const size_t bT = (size_t)b * Tc;
  const int hoff = h * HDc;

  v8f z = {0.f, 0.f, 0.f, 0.f, 0.f, 0.f, 0.f, 0.f};

  // Q as two A-fragments (hd 0..31, 32..63); rows have hd contiguous in memory
  v16bf aq[2];
#pragma unroll
  for (int c = 0; c < 2; ++c) {
    const unsigned short* pr =
        q + (bT + qt * 16 + l15) * Ec + hoff + c * 32 + ((lane & 16) ? 16 : 0);
    aq[c] = frag_from(*(const uint4*)pr, *(const uint4*)(pr + 8));
  }

  v8f o[4];                                 // O tile 16 x 64 (4 hd subtiles)
#pragma unroll
  for (int t = 0; t < 4; ++t) o[t] = z;
  float mrun[8], lrun[8];
#pragma unroll
  for (int v = 0; v < 8; ++v) { mrun[v] = -1e30f; lrun[v] = 0.0f; }

  const int qmax = qt * 16 + 15;
  for (int j = 0; j <= qmax; j += 32) {     // 32 kv positions per step
    // ---- S = Q * K^T (scaled) : two 16x16 f32 tiles -----------------------
    v8f s[2];
#pragma unroll
    for (int jn = 0; jn < 2; ++jn) {
      v8f a = z;
#pragma unroll
      for (int c = 0; c < 2; ++c) {
        const unsigned short* pr = k + (bT + j + jn * 16 + l15) * Ec + hoff +
                                   c * 32 + ((lane & 16) ? 16 : 0);
        v16bf bk = frag_from(*(const uint4*)pr, *(const uint4*)(pr + 8));
        a = wmma_bf16(aq[c], bk, a);
      }
      s[jn] = a;
    }
    const bool needMask = (j + 31 > qt * 16);
#pragma unroll
    for (int jn = 0; jn < 2; ++jn) {
#pragma unroll
      for (int v = 0; v < 8; ++v) {
        float sv = s[jn][v] * 0.125f;       // HD^-0.5
        if (needMask) {
          int kv   = j + jn * 16 + l15;
          int qrow = qt * 16 + v + m8;
          if (kv > qrow) sv = -1e30f;
        }
        s[jn][v] = sv;
      }
    }
    // ---- online softmax (rows split: lanes 0-15 -> v, lanes 16-31 -> v+8) --
#pragma unroll
    for (int v = 0; v < 8; ++v) {
      float mx = fmaxf(s[0][v], s[1][v]);
#pragma unroll
      for (int d = 8; d >= 1; d >>= 1) mx = fmaxf(mx, __shfl_xor(mx, d, 16));
      float mnew = fmaxf(mrun[v], mx);
      float fs = __expf(mrun[v] - mnew);
      mrun[v] = mnew;
      float p0 = __expf(s[0][v] - mnew);
      float p1 = __expf(s[1][v] - mnew);
      float rs = p0 + p1;
#pragma unroll
      for (int d = 8; d >= 1; d >>= 1) rs += __shfl_xor(rs, d, 16);
      lrun[v] = lrun[v] * fs + rs;
#pragma unroll
      for (int t = 0; t < 4; ++t) o[t][v] *= fs;
      int prow = v + m8;
      Pbuf[w][prow * 32 + l15]      = f2bf(p0);
      Pbuf[w][prow * 32 + 16 + l15] = f2bf(p1);
    }
    // wave-private LDS round trip; waves diverge so no block barrier allowed.
    // DS ops are in-order within a wave; force HW wait + compiler fence:
    asm volatile("s_wait_dscnt 0" ::: "memory");
    const unsigned short* pp = &Pbuf[w][l15 * 32 + ((lane & 16) ? 16 : 0)];
    v16bf ap = frag_from(*(const uint4*)pp, *(const uint4*)(pp + 8));

    // ---- O += P * V --------------------------------------------------------
    const int kb = (lane & 16) ? 16 : 0;
#pragma unroll
    for (int t4 = 0; t4 < 4; ++t4) {
      const unsigned short* vp = vv + (bT + j + kb) * Ec + hoff + t4 * 16 + l15;
      v16u vu;
#pragma unroll
      for (int t = 0; t < 16; ++t) vu[t] = vp[(size_t)t * Ec];
      o[t4] = wmma_bf16(ap, __builtin_bit_cast(v16bf, vu), o[t4]);
    }
  }
  // ---- epilogue: O / l, store bf16 -----------------------------------------
#pragma unroll
  for (int v = 0; v < 8; ++v) {
    float inv = 1.0f / lrun[v];
    unsigned short* orow = att + (bT + qt * 16 + v + m8) * Ec + hoff;
#pragma unroll
    for (int t4 = 0; t4 < 4; ++t4) orow[t4 * 16 + l15] = f2bf(o[t4][v] * inv);
  }
}

// ---- orchestration ---------------------------------------------------------
extern "C" void kernel_launch(void* const* d_in, const int* in_sizes, int n_in,
                              void* d_out, int out_size, void* d_ws, size_t ws_size,
                              hipStream_t stream) {
  (void)in_sizes; (void)n_in; (void)out_size; (void)ws_size;
  const int*   idx  = (const int*)d_in[0];
  const float* tok  = (const float*)d_in[1];
  const float* pos  = (const float*)d_in[2];
  const float* Wq   = (const float*)d_in[3];
  const float* Wk   = (const float*)d_in[4];
  const float* Wv   = (const float*)d_in[5];
  const float* Wo   = (const float*)d_in[6];
  const float* bo   = (const float*)d_in[7];
  const float* ln1s = (const float*)d_in[8];
  const float* ln1b = (const float*)d_in[9];
  const float* ln2s = (const float*)d_in[10];
  const float* ln2b = (const float*)d_in[11];
  const float* W1   = (const float*)d_in[12];
  const float* b1   = (const float*)d_in[13];
  const float* W2   = (const float*)d_in[14];
  const float* b2   = (const float*)d_in[15];
  const float* lnfs = (const float*)d_in[16];
  const float* lnfb = (const float*)d_in[17];
  const float* Wlm  = (const float*)d_in[18];
  const float* blm  = (const float*)d_in[19];

  char* ws = (char*)d_ws;
  size_t off = 0;
  auto take = [&](size_t bytes) -> void* {
    void* p = (void*)(ws + off);
    off += (bytes + 255) & ~(size_t)255;
    return p;
  };
  float*          x    = (float*)take((size_t)Mc * Ec * 4);
  unsigned short* h    = (unsigned short*)take((size_t)Mc * Ec * 2);
  unsigned short* qb   = (unsigned short*)take((size_t)Mc * Ec * 2);
  unsigned short* kb   = (unsigned short*)take((size_t)Mc * Ec * 2);
  unsigned short* vb   = (unsigned short*)take((size_t)Mc * Ec * 2);
  unsigned short* atb  = (unsigned short*)take((size_t)Mc * Ec * 2);
  unsigned short* mid  = (unsigned short*)take((size_t)Mc * 4 * Ec * 2);
  unsigned short* wtEE = (unsigned short*)take((size_t)Ec * Ec * 2);
  unsigned short* wt1  = (unsigned short*)take((size_t)Ec * 4 * Ec * 2);
  unsigned short* wt2  = (unsigned short*)take((size_t)4 * Ec * Ec * 2);
  unsigned short* wtLM = (unsigned short*)take((size_t)Ec * Vc * 2);

  const dim3 blk(256);
  const dim3 gEE(Ec / 128, Mc / 128);          // (4, 32)
  const dim3 gF1(4 * Ec / 128, Mc / 128);      // (16, 32)
  const dim3 gLM(Vc / 128, Mc / 128);          // (250, 32)
  const dim3 cEE(Ec / 32, Ec / 32);
  const dim3 cW1(4 * Ec / 32, Ec / 32);
  const dim3 cW2(Ec / 32, 4 * Ec / 32);
  const dim3 cLM(Vc / 32, Ec / 32);

  embed_kernel<<<(Mc * Ec / 4) / 256, blk, 0, stream>>>(idx, tok, pos, x);

  for (int l = 0; l < Lc; ++l) {
    const size_t oEE = (size_t)l * Ec * Ec, oF = (size_t)l * Ec * 4 * Ec;
    ln_kernel<<<Mc / 8, blk, 0, stream>>>(x, ln1s + l * Ec, ln1b + l * Ec, h);
    convT_kernel<<<cEE, blk, 0, stream>>>(Wq + oEE, wtEE, Ec, Ec);
    gemm_kernel<false, false, false, false, true><<<gEE, blk, 0, stream>>>(
        h, wtEE, nullptr, nullptr, nullptr, qb, Mc, Ec, Ec);
    convT_kernel<<<cEE, blk, 0, stream>>>(Wk + oEE, wtEE, Ec, Ec);
    gemm_kernel<false, false, false, false, true><<<gEE, blk, 0, stream>>>(
        h, wtEE, nullptr, nullptr, nullptr, kb, Mc, Ec, Ec);
    convT_kernel<<<cEE, blk, 0, stream>>>(Wv + oEE, wtEE, Ec, Ec);
    gemm_kernel<false, false, false, false, true><<<gEE, blk, 0, stream>>>(
        h, wtEE, nullptr, nullptr, nullptr, vb, Mc, Ec, Ec);

    attn_kernel<<<(Bc * Hc * (Tc / 16)) / 8, blk, 0, stream>>>(qb, kb, vb, atb);

    convT_kernel<<<cEE, blk, 0, stream>>>(Wo + oEE, wtEE, Ec, Ec);
    gemm_kernel<true, false, true, true, false><<<gEE, blk, 0, stream>>>(
        atb, wtEE, bo + l * Ec, x, x, nullptr, Mc, Ec, Ec);

    ln_kernel<<<Mc / 8, blk, 0, stream>>>(x, ln2s + l * Ec, ln2b + l * Ec, h);
    convT_kernel<<<cW1, blk, 0, stream>>>(W1 + oF, wt1, Ec, 4 * Ec);
    gemm_kernel<true, true, false, false, true><<<gF1, blk, 0, stream>>>(
        h, wt1, b1 + l * 4 * Ec, nullptr, nullptr, mid, Mc, 4 * Ec, Ec);
    convT_kernel<<<cW2, blk, 0, stream>>>(W2 + oF, wt2, 4 * Ec, Ec);
    gemm_kernel<true, false, true, true, false><<<gEE, blk, 0, stream>>>(
        mid, wt2, b2 + l * Ec, x, x, nullptr, Mc, Ec, 4 * Ec);
  }

  ln_kernel<<<Mc / 8, blk, 0, stream>>>(x, lnfs, lnfb, h);
  convT_kernel<<<cLM, blk, 0, stream>>>(Wlm, wtLM, Ec, Vc);
  gemm_kernel<true, false, false, true, false><<<gLM, blk, 0, stream>>>(
      h, wtLM, blm, nullptr, (float*)d_out, nullptr, Mc, Vc, Ec);
}
